// SelfAtten_34076270527142
// MI455X (gfx1250) — matrix operand order repaired and checked by reading, not compiled, not run on hardware
//
#include <hip/hip_runtime.h>
#include <hip/hip_bf16.h>

typedef __attribute__((ext_vector_type(16))) _Float16 v16h;
typedef __attribute__((ext_vector_type(8)))  _Float16 v8h;
typedef __attribute__((ext_vector_type(8)))  float    v8f;

#define B_  4
#define D_  128
#define K_  64
#define N_  4096

__device__ __forceinline__ v8f wmma_f32_f16(v16h a, v16h b, v8f c) {
    // (neg_a, A, neg_b, B, c_mod, C, reuse_a, reuse_b)
    return __builtin_amdgcn_wmma_f32_16x16x32_f16(false, a, false, b, (short)0, c, false, false);
}

// ---------------------------------------------------------------------------
// Stage 0: f32 -> f16 convert
// ---------------------------------------------------------------------------
__global__ void cvt_f32_f16_kernel(const float* __restrict__ src,
                                   _Float16* __restrict__ dst, int n) {
    int i = blockIdx.x * blockDim.x + threadIdx.x;
    if (i < n) dst[i] = (_Float16)src[i];
}

// ---------------------------------------------------------------------------
// Stage 1: fused QKV projection.
// One wave per (b, 16-row n-tile). A = x^T tile (16 n x 128 d), B = W^T.
// q,k -> [B][N][K_] f16 ; v -> [B][D][N] f16.
// ---------------------------------------------------------------------------
__global__ __launch_bounds__(32) void qkv_proj_kernel(
    const _Float16* __restrict__ xh,   // [B][D][N]
    const _Float16* __restrict__ Wqh,  // [K][D]
    const float*    __restrict__ bq,   // [K]
    const _Float16* __restrict__ Wkh,  // [K][D]
    const float*    __restrict__ bk,   // [K]
    const _Float16* __restrict__ Wvh,  // [D][D]
    const float*    __restrict__ bv,   // [D]
    _Float16* __restrict__ qh,         // [B][N][K]
    _Float16* __restrict__ kh,         // [B][N][K]
    _Float16* __restrict__ vh)         // [B][D][N]
{
    const int blk  = blockIdx.x;            // b * (N/16) + ntile
    const int b    = blk / (N_ / 16);
    const int nt   = blk % (N_ / 16);
    const int lane = threadIdx.x;           // 0..31
    const int lo   = lane & 15;
    const int hi   = lane >> 4;
    const int nrow = nt * 16 + lo;          // A-matrix row M = lo

    // A-fragments of x^T: one per 32-wide d-chunk (D=128 -> 4 chunks).
    // A 16x32 fragment in the contiguous k order used here: element e holds K = hi*16 + e.
    v16h af[4];
#pragma unroll
    for (int dc = 0; dc < 4; ++dc) {
#pragma unroll
        for (int e = 0; e < 16; ++e) {
            int d = dc * 32 + hi * 16 + e;
            af[dc][e] = xh[((size_t)b * D_ + d) * (size_t)N_ + nrow];
        }
    }

    // ---- Q and K projections: 4 column tiles of 16 channels each ----
#pragma unroll
    for (int ct = 0; ct < 4; ++ct) {
        const int kcol = ct * 16 + lo;      // B/C column = lane lo
        v8f accq = {}; v8f acck = {};
#pragma unroll
        for (int dc = 0; dc < 4; ++dc) {
            // B 32x16, same contiguous k order as A: element e -> K = hi*16 + e (contiguous row read)
            v16h bfq = *(const v16h*)(Wqh + (size_t)kcol * D_ + dc * 32 + hi * 16);
            accq = wmma_f32_f16(af[dc], bfq, accq);
            v16h bfk = *(const v16h*)(Wkh + (size_t)kcol * D_ + dc * 32 + hi * 16);
            acck = wmma_f32_f16(af[dc], bfk, acck);
        }
        const float biasq = bq[kcol];
        const float biask = bk[kcol];
#pragma unroll
        for (int r = 0; r < 8; ++r) {       // C/D: row M = r + hi*8
            const int n = nt * 16 + hi * 8 + r;
            qh[((size_t)b * N_ + n) * K_ + kcol] = (_Float16)(accq[r] + biasq);
            kh[((size_t)b * N_ + n) * K_ + kcol] = (_Float16)(acck[r] + biask);
        }
    }

    // ---- V projection: 8 column tiles of 16 channels ----
#pragma unroll
    for (int ct = 0; ct < 8; ++ct) {
        const int dcol = ct * 16 + lo;
        v8f acc = {};
#pragma unroll
        for (int dc = 0; dc < 4; ++dc) {
            v16h bf = *(const v16h*)(Wvh + (size_t)dcol * D_ + dc * 32 + hi * 16);
            acc = wmma_f32_f16(af[dc], bf, acc);
        }
        const float bias = bv[dcol];
        v8h pack;
#pragma unroll
        for (int r = 0; r < 8; ++r) pack[r] = (_Float16)(acc[r] + bias);
        // vh[b][dcol][n], n = nt*16 + hi*8 + r : contiguous 8 halves
        *(v8h*)(vh + ((size_t)b * D_ + dcol) * (size_t)N_ + nt * 16 + hi * 8) = pack;
    }
}

// ---------------------------------------------------------------------------
// Stage 2: fused flash attention. One wave per (b, 16-query-row tile).
// Per 32-key step: 4 WMMAs (energy) + online softmax + 8 WMMAs (P·V^T).
// Never materializes the N x N attention matrix.
// ---------------------------------------------------------------------------
__global__ __launch_bounds__(32) void flash_attn_kernel(
    const _Float16* __restrict__ qh,   // [B][N][K]
    const _Float16* __restrict__ kh,   // [B][N][K]
    const _Float16* __restrict__ vh,   // [B][D][N]
    float* __restrict__ out)           // [B][D][N]
{
    __shared__ _Float16 lds_p[16 * 32];     // P tile staging (C-layout -> A-layout)

    const int blk  = blockIdx.x;
    const int b    = blk / (N_ / 16);
    const int nt   = blk % (N_ / 16);
    const int lane = threadIdx.x;
    const int lo   = lane & 15;
    const int hi   = lane >> 4;

    // Q A-fragments, resident for the whole key loop (K=64 -> 2 chunks).
    v16h qf[2];
    {
        const _Float16* qrow = qh + ((size_t)b * N_ + nt * 16 + lo) * K_;
#pragma unroll
        for (int kc = 0; kc < 2; ++kc) {
            v8h l0 = *(const v8h*)(qrow + kc * 32 + hi * 16);       // K = hi*16  + 0..7
            v8h l1 = *(const v8h*)(qrow + kc * 32 + hi * 16 + 8);   // K = hi*16+8+ 0..7
            qf[kc] = __builtin_shufflevector(l0, l1,
                      0,1,2,3,4,5,6,7,8,9,10,11,12,13,14,15);
        }
    }

    v8f oacc[8];
#pragma unroll
    for (int t = 0; t < 8; ++t) oacc[t] = (v8f){};
    float mrow[8], lrow[8];
#pragma unroll
    for (int r = 0; r < 8; ++r) { mrow[r] = -1e30f; lrow[r] = 0.0f; }

    for (int m0 = 0; m0 < N_; m0 += 32) {
        // ---- energy tiles: cols m0..m0+15 (e0), m0+16..m0+31 (e1) ----
        v8f e0 = {}, e1 = {};
        {
            const _Float16* kr0 = kh + ((size_t)b * N_ + m0 + lo) * K_;
            const _Float16* kr1 = kh + ((size_t)b * N_ + m0 + 16 + lo) * K_;
#pragma unroll
            for (int kc = 0; kc < 2; ++kc) {
                v16h b0 = *(const v16h*)(kr0 + kc * 32 + hi * 16);
                e0 = wmma_f32_f16(qf[kc], b0, e0);
                v16h b1 = *(const v16h*)(kr1 + kc * 32 + hi * 16);
                e1 = wmma_f32_f16(qf[kc], b1, e1);
            }
        }

        // ---- online softmax (rows live in element index + hi half) ----
#pragma unroll
        for (int r = 0; r < 8; ++r) {
            float t = fmaxf(e0[r], e1[r]);
#pragma unroll
            for (int m = 1; m < 16; m <<= 1)
                t = fmaxf(t, __shfl_xor(t, m, 32));     // reduce within 16-lane half
            const float mn = fmaxf(mrow[r], t);
            const float sc = __expf(mrow[r] - mn);
            mrow[r] = mn;
            const float p0 = __expf(e0[r] - mn);
            const float p1 = __expf(e1[r] - mn);
            e0[r] = p0; e1[r] = p1;
            float s = p0 + p1;
#pragma unroll
            for (int m = 1; m < 16; m <<= 1)
                s += __shfl_xor(s, m, 32);
            lrow[r] = lrow[r] * sc + s;
#pragma unroll
            for (int t2 = 0; t2 < 8; ++t2) oacc[t2][r] *= sc;
        }

        // ---- stage P to LDS in (row, key) order; in-wave LDS is in-order ----
#pragma unroll
        for (int r = 0; r < 8; ++r) {
            lds_p[(hi * 8 + r) * 32 + lo]      = (_Float16)e0[r];
            lds_p[(hi * 8 + r) * 32 + 16 + lo] = (_Float16)e1[r];
        }
        __builtin_amdgcn_wave_barrier();

        // ---- read P back as a 16x32 A-fragment ----
        v16h pf;
        {
            const _Float16* prow = lds_p + lo * 32;
            v8h l0 = *(const v8h*)(prow + hi * 16);
            v8h l1 = *(const v8h*)(prow + hi * 16 + 8);
            pf = __builtin_shufflevector(l0, l1,
                  0,1,2,3,4,5,6,7,8,9,10,11,12,13,14,15);
        }
        __builtin_amdgcn_wave_barrier();

        // ---- O[n, d] += P[n, m] * V^T[m, d] ; V^T rows contiguous in vh[d][m] ----
#pragma unroll
        for (int t = 0; t < 8; ++t) {
            const int dcol = t * 16 + lo;
            v16h bf = *(const v16h*)(vh + ((size_t)b * D_ + dcol) * (size_t)N_ + m0 + hi * 16);
            oacc[t] = wmma_f32_f16(pf, bf, oacc[t]);
        }
    }

    // ---- normalize and store out[b][d][n] (8 contiguous floats per lane/tile) ----
#pragma unroll
    for (int t = 0; t < 8; ++t) {
        const int dcol = t * 16 + lo;
        float* op = out + ((size_t)b * D_ + dcol) * (size_t)N_ + nt * 16 + hi * 8;
#pragma unroll
        for (int r = 0; r < 8; ++r)
            op[r] = oacc[t][r] / lrow[r];
    }
}

// ---------------------------------------------------------------------------
// Host launcher
// ---------------------------------------------------------------------------
extern "C" void kernel_launch(void* const* d_in, const int* in_sizes, int n_in,
                              void* d_out, int out_size, void* d_ws, size_t ws_size,
                              hipStream_t stream) {
    const float* x  = (const float*)d_in[0];   // [B][D][N]
    const float* Wq = (const float*)d_in[1];   // [K][D]
    const float* bq = (const float*)d_in[2];   // [K]
    const float* Wk = (const float*)d_in[3];   // [K][D]
    const float* bk = (const float*)d_in[4];   // [K]
    const float* Wv = (const float*)d_in[5];   // [D][D]
    const float* bv = (const float*)d_in[6];   // [D]
    float* out = (float*)d_out;                // [B][D][N]

    // Workspace layout (bytes):
    char* ws = (char*)d_ws;
    _Float16* xh  = (_Float16*)(ws);                         // 4 MB  [B][D][N]
    _Float16* qh  = (_Float16*)(ws + 4194304);               // 2 MB  [B][N][K]
    _Float16* khp = (_Float16*)(ws + 6291456);               // 2 MB  [B][N][K]
    _Float16* vhp = (_Float16*)(ws + 8388608);               // 4 MB  [B][D][N]
    _Float16* Wqh = (_Float16*)(ws + 12582912);              // 16 KB
    _Float16* Wkh = (_Float16*)(ws + 12599296);              // 16 KB
    _Float16* Wvh = (_Float16*)(ws + 12615680);              // 32 KB

    const int nx = B_ * D_ * N_;
    const int nwqk = K_ * D_;
    const int nwv = D_ * D_;
    cvt_f32_f16_kernel<<<(nx   + 255) / 256, 256, 0, stream>>>(x,  xh,  nx);
    cvt_f32_f16_kernel<<<(nwqk + 255) / 256, 256, 0, stream>>>(Wq, Wqh, nwqk);
    cvt_f32_f16_kernel<<<(nwqk + 255) / 256, 256, 0, stream>>>(Wk, Wkh, nwqk);
    cvt_f32_f16_kernel<<<(nwv  + 255) / 256, 256, 0, stream>>>(Wv, Wvh, nwv);

    const int nblocks = B_ * (N_ / 16);      // 1024
    qkv_proj_kernel<<<nblocks, 32, 0, stream>>>(xh, Wqh, bq, Wkh, bk, Wvh, bv,
                                                qh, khp, vhp);
    flash_attn_kernel<<<nblocks, 32, 0, stream>>>(qh, khp, vhp, out);
}
